// LSTM_51049981280962
// MI455X (gfx1250) — matrix operand order repaired and checked by reading, not compiled, not run on hardware
//
#include <hip/hip_runtime.h>

// LSTM: B=32, T=512, D=512, H=1024 (PyTorch gate order i,f,g,o) -> h_T [B,H] f32.
//
// MI455X (gfx1250) strategy:
//  1) down-convert x, W_ih, W_hh to bf16 (matmul operands only; additive terms
//     and accumulation stay f32).
//  2) one big parallel WMMA GEMM: xg[t][b][n] = x·W_ihT + b_ih + b_hh (69 GFLOP,
//     runs at full device throughput, off the critical path).
//  3) ONE persistent kernel for the 512-step recurrence (64 blocks, trivially
//     co-resident). Per step: each block computes a 32x16 chunk of all four
//     gates (4 waves = 4 gates, each wave does both 16-row M-tiles so every
//     W_hh B-fragment feeds 2 WMMAs), K-loop of v_wmma_f32_16x16x32_bf16
//     seeded from the f32 xg tile, gates exchanged via LDS, cell state c held
//     entirely in registers (block owns its columns for the whole sequence),
//     h double-buffered in bf16 through global, grid-wide barrier per step via
//     agent-scope atomics + s_sleep spin. W_hh(bf16)=8MB stays L2-resident.

typedef __attribute__((ext_vector_type(16))) __bf16 v16bf;
typedef __attribute__((ext_vector_type(8)))  __bf16 v8bf;
typedef __attribute__((ext_vector_type(8)))  float  v8f;

#define B_   32
#define T_   512
#define D_   512
#define H_   1024
#define G4H  (4 * H_)      // 4096
#define NBLK (H_ / 16)     // 64 persistent blocks

// ---------------------------------------------------------------- utilities

__global__ void f32_to_bf16_kernel(const float* __restrict__ src,
                                   __bf16* __restrict__ dst, int n) {
  int i = blockIdx.x * blockDim.x + threadIdx.x;
  int stride = gridDim.x * blockDim.x;
  for (; i < n; i += stride) dst[i] = (__bf16)src[i];
}

__global__ void init_state_kernel(__bf16* __restrict__ h0,
                                  unsigned* __restrict__ counter) {
  int i = blockIdx.x * blockDim.x + threadIdx.x;
  if (i < B_ * H_) h0[i] = (__bf16)0.0f;
  if (i == 0) *counter = 0u;   // reset grid-barrier counter every call
}

__device__ __forceinline__ float fast_sigmoid(float x) {
  return 1.0f / (1.0f + __expf(-x));
}
__device__ __forceinline__ float fast_tanh(float x) {
  return 2.0f * fast_sigmoid(2.0f * x) - 1.0f;
}

// A fragment (16x32 bf16, MxK) from row-major [*, ld] at (m0, k0).
// lanes 0-15: row m0+lane, K k0+0..7 (v0-3) and k0+16..23 (v4-7);
// lanes 16-31: row m0+lane-16, K k0+8..15 and k0+24..31.
__device__ __forceinline__ v16bf load_frag_a(const __bf16* __restrict__ base,
                                             int ld, int m0, int k0, int lane) {
  int half = lane >> 4;
  int r    = lane & 15;
  const __bf16* p = base + (size_t)(m0 + r) * ld + k0 + half * 8;
  v8bf lo = *(const v8bf*)(p);        // 16B aligned
  v8bf hi = *(const v8bf*)(p + 16);   // 16B aligned
  return __builtin_shufflevector(lo, hi, 0, 1, 2, 3, 4, 5, 6, 7,
                                         8, 9, 10, 11, 12, 13, 14, 15);
}

// B fragment (32x16 bf16, KxN), B[k][n] = W[n0+n][k0+k], W row-major [*, ldk].
// Column n of B == contiguous row n0+n of W; same K order as A fragment:
// lanes 0-15: K k0+0..7 and k0+16..23; lanes 16-31: K k0+8..15 and k0+24..31.
__device__ __forceinline__ v16bf load_frag_b(const __bf16* __restrict__ w,
                                             int ldk, int n0, int k0, int lane) {
  int half = lane >> 4;
  int col  = lane & 15;
  const __bf16* p = w + (size_t)(n0 + col) * ldk + k0 + half * 8;
  v8bf lo = *(const v8bf*)(p);        // 16B aligned
  v8bf hi = *(const v8bf*)(p + 16);   // 16B aligned
  return __builtin_shufflevector(lo, hi, 0, 1, 2, 3, 4, 5, 6, 7,
                                         8, 9, 10, 11, 12, 13, 14, 15);
}

// ------------------------------------------------ phase 1: input projection
// xg[t][b][n] = sum_d x[b][t][d]*W_ih[n][d] + b_ih[n] + b_hh[n]
// grid: (4H/64, B/16, T), block: 128 (4 waves; wave w -> 16-col subtile)
__global__ void xproj_wmma_kernel(const __bf16* __restrict__ xb,   // [B][T][D]
                                  const __bf16* __restrict__ wih,  // [4H][D]
                                  const float* __restrict__ b_ih,  // [4H]
                                  const float* __restrict__ b_hh,  // [4H]
                                  float* __restrict__ xg)          // [T][B][4H]
{
  const int lane = threadIdx.x & 31;
  const int wave = threadIdx.x >> 5;
  const int t    = blockIdx.z;
  const int m0   = blockIdx.y * 16;
  const int n0   = (blockIdx.x * 4 + wave) * 16;
  const int col  = lane & 15;
  const int rbase = (lane >> 4) * 8;

  float bias = b_ih[n0 + col] + b_hh[n0 + col];
  v8f acc;
#pragma unroll
  for (int e = 0; e < 8; ++e) acc[e] = bias;

#pragma unroll 4
  for (int k0 = 0; k0 < D_; k0 += 32) {
    int half = lane >> 4, r = lane & 15;
    const __bf16* pa = xb + ((size_t)(m0 + r) * T_ + t) * D_ + k0 + half * 8;
    v8bf lo = *(const v8bf*)(pa);
    v8bf hi = *(const v8bf*)(pa + 16);
    v16bf a = __builtin_shufflevector(lo, hi, 0, 1, 2, 3, 4, 5, 6, 7,
                                              8, 9, 10, 11, 12, 13, 14, 15);
    v16bf b = load_frag_b(wih, D_, n0, k0, lane);
    acc = __builtin_amdgcn_wmma_f32_16x16x32_bf16(false, a, false, b,
                                                  (short)0, acc, false, false);
  }

  float* outp = xg + (size_t)t * B_ * G4H;
#pragma unroll
  for (int e = 0; e < 8; ++e)
    outp[(size_t)(m0 + rbase + e) * G4H + (n0 + col)] = acc[e];
}

// ------------------------------------------------ phase 2: persistent recurrence

__device__ __forceinline__ void grid_barrier(unsigned* __restrict__ counter,
                                             unsigned target) {
  __threadfence();          // make this block's h stores visible device-wide
  __syncthreads();          // all threads of block done storing
  if (threadIdx.x == 0) {
    __hip_atomic_fetch_add(counter, 1u, __ATOMIC_ACQ_REL,
                           __HIP_MEMORY_SCOPE_AGENT);
    while (__hip_atomic_load(counter, __ATOMIC_ACQUIRE,
                             __HIP_MEMORY_SCOPE_AGENT) < target) {
      __builtin_amdgcn_s_sleep(1);
    }
  }
  __syncthreads();          // release whole block past the barrier
}

// grid: NBLK=64 blocks, 128 threads (wave w = gate w: i,f,g,o).
// Block owns gate columns h0..h0+15 for ALL 32 batch rows, all timesteps.
__global__ void lstm_persistent_kernel(__bf16* __restrict__ hbuf0,  // [B][H]
                                       __bf16* __restrict__ hbuf1,  // [B][H]
                                       const __bf16* __restrict__ whh, // [4H][H]
                                       const float* __restrict__ xg,   // [T][B][4H]
                                       float* __restrict__ h_out,      // [B][H]=d_out
                                       unsigned* __restrict__ counter)
{
  __shared__ float gsm[4][32][16];   // 4 gates x 32 rows x 16 cols = 8 KB

  const int lane  = threadIdx.x & 31;
  const int wave  = threadIdx.x >> 5;        // gate index
  const int h0    = blockIdx.x * 16;
  const int n0    = wave * H_ + h0;          // column in [0, 4H)
  const int col   = lane & 15;
  const int rbase = (lane >> 4) * 8;

  // per-thread persistent cell state: 4 cells (row rr, col cc)
  float creg[4];
#pragma unroll
  for (int q = 0; q < 4; ++q) creg[q] = 0.0f;

  for (int t = 0; t < T_; ++t) {
    const __bf16* hin  = (t & 1) ? hbuf1 : hbuf0;
    __bf16*       hout = (t & 1) ? hbuf0 : hbuf1;
    const float*  xg_t = xg + (size_t)t * B_ * G4H;

    // seed both M-tile accumulators with the f32 projection (C-layout match)
    v8f acc0, acc1;
#pragma unroll
    for (int e = 0; e < 8; ++e) {
      acc0[e] = xg_t[(size_t)(rbase + e) * G4H + (n0 + col)];
      acc1[e] = xg_t[(size_t)(16 + rbase + e) * G4H + (n0 + col)];
    }

#pragma unroll 4
    for (int k0 = 0; k0 < H_; k0 += 32) {
      v16bf b  = load_frag_b(whh, H_, n0, k0, lane);      // shared by both tiles
      v16bf a0 = load_frag_a(hin, H_, 0,  k0, lane);
      v16bf a1 = load_frag_a(hin, H_, 16, k0, lane);
      acc0 = __builtin_amdgcn_wmma_f32_16x16x32_bf16(false, a0, false, b,
                                                     (short)0, acc0, false, false);
      acc1 = __builtin_amdgcn_wmma_f32_16x16x32_bf16(false, a1, false, b,
                                                     (short)0, acc1, false, false);
    }

#pragma unroll
    for (int e = 0; e < 8; ++e) {
      gsm[wave][rbase + e][col]      = acc0[e];
      gsm[wave][16 + rbase + e][col] = acc1[e];
    }
    __syncthreads();

    // 512 cells per block, 128 threads -> 4 cells each (fixed mapping so
    // creg[q] tracks the same cell across all timesteps)
#pragma unroll
    for (int q = 0; q < 4; ++q) {
      int idx = threadIdx.x + q * 128;
      int r  = idx >> 4;          // batch row 0..31
      int cc = idx & 15;          // column within tile
      float i_ = fast_sigmoid(gsm[0][r][cc]);
      float f_ = fast_sigmoid(gsm[1][r][cc]);
      float g_ = fast_tanh   (gsm[2][r][cc]);
      float o_ = fast_sigmoid(gsm[3][r][cc]);
      float cn = f_ * creg[q] + i_ * g_;
      creg[q] = cn;
      float hn = o_ * fast_tanh(cn);
      size_t off = (size_t)r * H_ + (h0 + cc);
      hout[off] = (__bf16)hn;
      if (t == T_ - 1) h_out[off] = hn;
    }
    __syncthreads();   // protect gsm reuse next iteration

    grid_barrier(counter, (unsigned)(NBLK * (t + 1)));
  }
}

// ---------------------------------------------------------------- launcher

extern "C" void kernel_launch(void* const* d_in, const int* in_sizes, int n_in,
                              void* d_out, int out_size, void* d_ws, size_t ws_size,
                              hipStream_t stream) {
  const float* x    = (const float*)d_in[0];   // [B][T][D]
  const float* W_ih = (const float*)d_in[1];   // [4H][D]
  const float* W_hh = (const float*)d_in[2];   // [4H][H]
  const float* b_ih = (const float*)d_in[3];   // [4H]
  const float* b_hh = (const float*)d_in[4];   // [4H]
  float* h_final = (float*)d_out;              // [B][H]

  char* ws = (char*)d_ws;
  size_t off = 0;
  float*  xg    = (float*)(ws + off);  off += (size_t)T_ * B_ * G4H * sizeof(float);   // 256 MiB
  __bf16* whh_b = (__bf16*)(ws + off); off += (size_t)G4H * H_ * sizeof(__bf16);       //   8 MiB
  __bf16* wih_b = (__bf16*)(ws + off); off += (size_t)G4H * D_ * sizeof(__bf16);       //   4 MiB
  __bf16* x_b   = (__bf16*)(ws + off); off += (size_t)B_ * T_ * D_ * sizeof(__bf16);   //  16 MiB
  __bf16* hbuf0 = (__bf16*)(ws + off); off += (size_t)B_ * H_ * sizeof(__bf16);
  __bf16* hbuf1 = (__bf16*)(ws + off); off += (size_t)B_ * H_ * sizeof(__bf16);
  unsigned* ctr = (unsigned*)(ws + off); off += 256;

  // precision down-converts for matmul operands
  f32_to_bf16_kernel<<<dim3(2048), 256, 0, stream>>>(W_hh, whh_b, G4H * H_);
  f32_to_bf16_kernel<<<dim3(1024), 256, 0, stream>>>(W_ih, wih_b, G4H * D_);
  f32_to_bf16_kernel<<<dim3(4096), 256, 0, stream>>>(x,    x_b,   B_ * T_ * D_);
  init_state_kernel<<<dim3((B_ * H_ + 255) / 256), 256, 0, stream>>>(hbuf0, ctr);

  // parallel input projection (all timesteps, full device throughput)
  xproj_wmma_kernel<<<dim3(G4H / 64, B_ / 16, T_), 128, 0, stream>>>(
      x_b, wih_b, b_ih, b_hh, xg);

  // single persistent kernel runs the entire 512-step recurrence
  lstm_persistent_kernel<<<dim3(NBLK), 128, 0, stream>>>(
      hbuf0, hbuf1, whh_b, xg, h_final, ctr);
}
